// PDAN_77086073028973
// MI455X (gfx1250) — matrix operand order repaired and checked by reading, not compiled, not run on hardware
//
#include <hip/hip_runtime.h>
#include <cstdint>
#include <cstddef>

// ---------------- model constants ----------------
#define L_   5
#define FM_  512
#define DIM_ 1024
#define NC_  157
#define NT_  10
#define NH_  4
#define B_   8
#define T_   4096
#define DH_  128           // FM/NH
#define SCALE_DH 0.08838834764831845f  // 1/sqrt(128)

typedef __attribute__((ext_vector_type(16))) __bf16 v16bf;
typedef __attribute__((ext_vector_type(8)))  __bf16 v8bf;
typedef __attribute__((ext_vector_type(8)))  float  v8f;

// Async global->LDS path (gfx1250): guarded so absence of the builtin cannot
// break the build (host pass also sees this as 0).
#if defined(__AMDGCN__) && defined(__has_builtin)
#if __has_builtin(__builtin_amdgcn_global_load_async_to_lds_b128) && \
    __has_builtin(__builtin_amdgcn_s_wait_asynccnt)
#define USE_ASYNC_COPY 1
#endif
#endif
#ifndef USE_ASYNC_COPY
#define USE_ASYNC_COPY 0
#endif

#if USE_ASYNC_COPY
// Builtin signature (from compiler diagnostic): param 1 is a global-AS pointer
// to 'int __attribute__((vector_size(16)))'. Use matching typed pointers.
typedef int v4i_ __attribute__((__vector_size__(16)));
typedef __attribute__((address_space(1))) v4i_ gv4i_t;
typedef __attribute__((address_space(3))) v4i_ lv4i_t;
#endif

// ---------------- elementwise conversion ----------------
__global__ void f32_to_bf16_k(const float* __restrict__ s, __bf16* __restrict__ d, size_t n) {
  size_t i = (size_t)blockIdx.x * blockDim.x + threadIdx.x;
  if (i < n) d[i] = (__bf16)s[i];
}

// ---------------- core WMMA GEMM ----------------
// Y[b] (M x N) = A (M x K) @ X[b] (K x N), bf16 inputs, f32 accumulate.
// Block tile 128x128, K-step 32, 8 waves (wave32): wave grid 4(M) x 2(N),
// each wave owns 32x64 = 2x4 tiles of v_wmma_f32_16x16x32_bf16.
#define BM 128
#define BN 128
#define BK 32

template<bool HAS_BIAS, bool HAS_RES, bool HAS_MASK, bool WF32, bool WBF16>
__global__ __launch_bounds__(256) void wmma_gemm(
    const __bf16* __restrict__ A,   // (M,K) row-major, K-contiguous
    const __bf16* __restrict__ X,   // (batch,K,N), N-contiguous
    const float* __restrict__ bias, // (M) or null
    const float* res,               // (batch,outRows,N) or null (may alias Yf)
    const float* __restrict__ mask, // (batch,N) or null
    float* Yf, __bf16* __restrict__ Yb,
    int M, int K, int N, int outRows, int rowOff)
{
  __shared__ __align__(16) __bf16 lA[2][BM * BK];   // [m][k]
  __shared__ __align__(16) __bf16 lB[2][BN * BK];   // transposed: [n][k]

  const int tid  = threadIdx.x;
  const int lane = tid & 31;
  const int wave = tid >> 5;
  const int wm   = wave & 3;       // 4 waves along M
  const int wn   = wave >> 2;      // 2 waves along N
  const int b    = blockIdx.z;
  const int mBase = blockIdx.y * BM;
  const int nBase = blockIdx.x * BN;

  const __bf16* Xb = X + (size_t)b * K * N;

  const v8f vzero = {0.f, 0.f, 0.f, 0.f, 0.f, 0.f, 0.f, 0.f};
  v8f acc[2][4];
#pragma unroll
  for (int i = 0; i < 2; ++i)
#pragma unroll
    for (int j = 0; j < 4; ++j) acc[i][j] = vzero;

  const int nK = K / BK;

  auto stage = [&](int ks, int buf) {
    const int k0 = ks * BK;
    // ---- A tile: 128 rows x 32 cols, 512 chunks of 8 bf16 (16B) ----
#pragma unroll
    for (int j = 0; j < 2; ++j) {
      const int c  = tid * 2 + j;
      const int r  = c >> 2;
      const int ko = (c & 3) * 8;
      const int gr = mBase + r;
      if (gr < M) {
#if USE_ASYNC_COPY
        __builtin_amdgcn_global_load_async_to_lds_b128(
            (gv4i_t*)(uintptr_t)(A + (size_t)gr * K + k0 + ko),
            (lv4i_t*)(unsigned)(uintptr_t)(&lA[buf][r * BK + ko]), 0, 0);
#else
        v8bf av = *(const v8bf*)(A + (size_t)gr * K + k0 + ko);
        *(v8bf*)&lA[buf][r * BK + ko] = av;
#endif
        if (k0 + BK < K)  // speculative prefetch of the k+2 tile
          __builtin_prefetch(A + (size_t)gr * K + k0 + BK + ko, 0, 1);
      } else {
#pragma unroll
        for (int e = 0; e < 8; ++e) lA[buf][r * BK + ko + e] = (__bf16)0.f;
      }
    }
    // ---- B tile: 32 k-rows x 128 n-cols, coalesced read + transposed LDS store ----
#pragma unroll
    for (int j = 0; j < 2; ++j) {
      const int c  = tid * 2 + j;
      const int kr = c >> 4;
      const int no = (c & 15) * 8;
      v8bf bv = *(const v8bf*)(Xb + (size_t)(k0 + kr) * N + nBase + no);
#pragma unroll
      for (int e = 0; e < 8; ++e) lB[buf][(no + e) * BK + kr] = bv[e];
      if (k0 + BK < K)
        __builtin_prefetch(Xb + (size_t)(k0 + BK + kr) * N + nBase + no, 0, 1);
    }
  };

  stage(0, 0);
#if USE_ASYNC_COPY
  __builtin_amdgcn_s_wait_asynccnt(0);
#endif
  __syncthreads();

  const int rsel = lane & 15;
  const int hsel = lane >> 4;

  for (int ks = 0; ks < nK; ++ks) {
    const int cur = ks & 1, nxt = cur ^ 1;
    if (ks + 1 < nK) stage(ks + 1, nxt);

    // A fragments: lane<16 -> K {0..7,16..23}; lane>=16 -> K {8..15,24..31}
    v16bf af[2];
#pragma unroll
    for (int mt = 0; mt < 2; ++mt) {
      const int row = wm * 32 + mt * 16 + rsel;
      v8bf lo = *(const v8bf*)&lA[cur][row * BK + hsel * 8];
      v8bf hi = *(const v8bf*)&lA[cur][row * BK + hsel * 8 + 16];
      af[mt] = __builtin_shufflevector(lo, hi, 0,1,2,3,4,5,6,7,8,9,10,11,12,13,14,15);
    }
    // B fragments: lane<16 -> col=lane, K {0..7,16..23}; lane>=16 -> col=lane-16, K {8..15,24..31}
    v16bf bfr[4];
#pragma unroll
    for (int nt = 0; nt < 4; ++nt) {
      const int col = wn * 64 + nt * 16 + rsel;
      v8bf lo = *(const v8bf*)&lB[cur][col * BK + hsel * 8];
      v8bf hi = *(const v8bf*)&lB[cur][col * BK + hsel * 8 + 16];
      bfr[nt] = __builtin_shufflevector(lo, hi, 0,1,2,3,4,5,6,7,8,9,10,11,12,13,14,15);
    }
#pragma unroll
    for (int mt = 0; mt < 2; ++mt)
#pragma unroll
      for (int nt = 0; nt < 4; ++nt)
        acc[mt][nt] = __builtin_amdgcn_wmma_f32_16x16x32_bf16(
            false, af[mt], false, bfr[nt], (short)0, acc[mt][nt], false, false);

    if (ks + 1 < nK) {
#if USE_ASYNC_COPY
      __builtin_amdgcn_s_wait_asynccnt(0);
#endif
      __syncthreads();
    }
  }

  // ---- epilogue: C/D layout -> VGPR r holds M=r (lanes<16) / M=8+r (lanes>=16)
#pragma unroll
  for (int mt = 0; mt < 2; ++mt)
#pragma unroll
    for (int nt = 0; nt < 4; ++nt) {
      const int col   = nBase + wn * 64 + nt * 16 + rsel;
      const int rbase = mBase + wm * 32 + mt * 16 + hsel * 8;
      float mval = 1.f;
      if (HAS_MASK) mval = mask[(size_t)b * N + col];
#pragma unroll
      for (int r = 0; r < 8; ++r) {
        const int row = rbase + r;
        if (row < M) {
          float v = acc[mt][nt][r];
          if (HAS_BIAS) v += bias[row];
          const size_t offo = ((size_t)b * outRows + rowOff + row) * N + col;
          if (HAS_RES) v += res[offo];
          if (HAS_MASK) v *= mval;
          if (WF32)  Yf[offo] = v;
          if (WBF16) Yb[offo] = (__bf16)v;
        }
      }
    }
}

// ---------------- summary-token query projection (done once) ----------------
// qsm[q][e] = tokens[q,:] . sm_in_w[e,:] + sm_in_b[e], e in [0,512)
__global__ __launch_bounds__(256) void qsm_proj_k(const float* __restrict__ tokens,
                                                  const float* __restrict__ w,
                                                  const float* __restrict__ bvec,
                                                  float* __restrict__ qsm) {
  const int q = blockIdx.x;
  for (int e = threadIdx.x; e < FM_; e += 256) {
    float s = bvec[e];
    const float* wr = w + (size_t)e * FM_;
    const float* tk = tokens + (size_t)q * FM_;
    for (int d = 0; d < FM_; ++d) s += tk[d] * wr[d];
    qsm[q * FM_ + e] = s;
  }
}

// ---------------- summary attention: 10 queries over T=4096 keys ----------------
// kv: (B,1024,T) bf16, rows 0..511 = k-proj, 512..1023 = v-proj. osm: (B,10,512) f32.
__global__ __launch_bounds__(256) void sm_attention_k(const float* __restrict__ qsm,
                                                      const __bf16* __restrict__ kv,
                                                      float* __restrict__ osm) {
  const int q = blockIdx.x, h = blockIdx.y, b = blockIdx.z;
  const int tid = threadIdx.x;
  __shared__ float sq[DH_];
  __shared__ float sp[T_];
  __shared__ float rbuf[256];

  if (tid < DH_) sq[tid] = qsm[q * FM_ + h * DH_ + tid] * SCALE_DH;
  __syncthreads();

  const __bf16* kb = kv + ((size_t)b * 1024 + h * DH_) * T_;
  float lmax = -1e30f;
  for (int t = tid; t < T_; t += 256) {
    float s = 0.f;
#pragma unroll 8
    for (int d = 0; d < DH_; ++d) s += sq[d] * (float)kb[(size_t)d * T_ + t];
    sp[t] = s;
    lmax = fmaxf(lmax, s);
  }
  rbuf[tid] = lmax; __syncthreads();
  for (int s = 128; s > 0; s >>= 1) { if (tid < s) rbuf[tid] = fmaxf(rbuf[tid], rbuf[tid + s]); __syncthreads(); }
  const float gmax = rbuf[0];
  __syncthreads();

  float lsum = 0.f;
  for (int t = tid; t < T_; t += 256) { float p = __expf(sp[t] - gmax); sp[t] = p; lsum += p; }
  rbuf[tid] = lsum; __syncthreads();
  for (int s = 128; s > 0; s >>= 1) { if (tid < s) rbuf[tid] += rbuf[tid + s]; __syncthreads(); }
  const float gsum = rbuf[0];
  __syncthreads();

  const __bf16* vb = kv + ((size_t)b * 1024 + FM_ + h * DH_) * T_;
  const int d = tid & 127, half = tid >> 7;
  float o = 0.f;
  for (int t = half * 2048; t < half * 2048 + 2048; ++t)
    o += sp[t] * (float)vb[(size_t)d * T_ + t];
  rbuf[tid] = o; __syncthreads();
  if (tid < DH_)
    osm[((size_t)(b * NT_ + q)) * FM_ + h * DH_ + tid] = (rbuf[tid] + rbuf[tid + 128]) / gsum;
}

// summary += osm @ sm_out_w^T + sm_out_b   (80 tiny rows)
__global__ __launch_bounds__(256) void smout_acc_k(const float* __restrict__ osm,
                                                   const float* __restrict__ w,
                                                   const float* __restrict__ bvec,
                                                   float* __restrict__ summary, int first) {
  const int bq = blockIdx.x;
  for (int e = threadIdx.x; e < FM_; e += 256) {
    float s = bvec[e];
    const float* o  = osm + (size_t)bq * FM_;
    const float* wr = w + (size_t)e * FM_;
    for (int d = 0; d < FM_; ++d) s += o[d] * wr[d];
    const size_t idx = (size_t)bq * FM_ + e;
    summary[idx] = (first ? 0.f : summary[idx]) + s;
  }
}

// ---------------- DAL 3-tap windowed softmax + ReLU ----------------
// qkv: (B,1536,T) bf16 (rows 0..511 q, 512..1023 k, 1024..1535 v); rel: (FM,3) f32
__global__ __launch_bounds__(256) void dal_elem_k(const __bf16* __restrict__ qkv,
                                                  const float* __restrict__ rel,
                                                  __bf16* __restrict__ hout, int dil) {
  const size_t idx = (size_t)blockIdx.x * 256 + threadIdx.x;
  if (idx >= (size_t)B_ * FM_ * T_) return;
  const int t = (int)(idx & (T_ - 1));
  const size_t r = idx >> 12;
  const int o = (int)(r & (FM_ - 1));
  const int b = (int)(r >> 9);
  const __bf16* base = qkv + (size_t)b * 1536 * T_;
  const float qv = (float)base[(size_t)o * T_ + t];
  float sc[3], vv[3];
#pragma unroll
  for (int tap = 0; tap < 3; ++tap) {
    const int ts = t + (tap - 1) * dil;
    const bool ok = (ts >= 0) && (ts < T_);
    const float kk = ok ? (float)base[(size_t)(FM_ + o) * T_ + ts] : 0.f;
    const float vs = ok ? (float)base[(size_t)(2 * FM_ + o) * T_ + ts] : 0.f;
    sc[tap] = qv * kk;
    vv[tap] = vs + rel[o * 3 + tap];
  }
  const float mx = fmaxf(sc[0], fmaxf(sc[1], sc[2]));
  const float e0 = __expf(sc[0] - mx), e1 = __expf(sc[1] - mx), e2 = __expf(sc[2] - mx);
  const float hsum = (e0 * vv[0] + e1 * vv[1] + e2 * vv[2]) / (e0 + e1 + e2);
  hout[((size_t)b * FM_ + o) * T_ + t] = (__bf16)fmaxf(hsum, 0.f);
}

// ---------------- cross-attention k/v projection from summary/NUM_LAYERS ----------------
// kvca[b][n][e] (e<512: k, e>=512: v) = (summary[b,n,:]/5) . ca_in_w[512+e,:]
__global__ __launch_bounds__(256) void ca_kv_k(const float* __restrict__ summary,
                                               const float* __restrict__ caw,
                                               float* __restrict__ kvca) {
  const int bn = blockIdx.x;
  for (int e = threadIdx.x; e < 2 * FM_; e += 256) {
    float s = 0.f;
    const float* sm = summary + (size_t)bn * FM_;
    const float* wr = caw + (size_t)e * FM_;
    for (int d = 0; d < FM_; ++d) s += (sm[d] * 0.2f) * wr[d];
    kvca[(size_t)bn * 2 * FM_ + e] = s;
  }
}

// ---------------- cross-attention over 10 summary tokens ----------------
__global__ __launch_bounds__(256) void ca_att_k(const __bf16* __restrict__ qca,
                                                const float* __restrict__ kvca,
                                                __bf16* __restrict__ oca) {
  const int b = blockIdx.y;
  const int t = blockIdx.x * 64 + (threadIdx.x & 63);
  const int h = threadIdx.x >> 6;
  const __bf16* qb = qca + ((size_t)b * FM_ + h * DH_) * T_;
  float sc[NT_];
  float mx = -1e30f;
#pragma unroll
  for (int n = 0; n < NT_; ++n) {
    float s = 0.f;
    const float* kk = kvca + ((size_t)(b * NT_ + n)) * (2 * FM_) + h * DH_;
    for (int d = 0; d < DH_; ++d) s += (float)qb[(size_t)d * T_ + t] * kk[d];
    s *= SCALE_DH;
    sc[n] = s;
    mx = fmaxf(mx, s);
  }
  float sum = 0.f;
#pragma unroll
  for (int n = 0; n < NT_; ++n) { sc[n] = __expf(sc[n] - mx); sum += sc[n]; }
  const float inv = 1.f / sum;
  for (int d = 0; d < DH_; ++d) {
    float o = 0.f;
#pragma unroll
    for (int n = 0; n < NT_; ++n)
      o += sc[n] * kvca[((size_t)(b * NT_ + n)) * (2 * FM_) + FM_ + h * DH_ + d];
    oca[((size_t)b * FM_ + h * DH_ + d) * T_ + t] = (__bf16)(o * inv);
  }
}

// ---------------- host orchestration ----------------
extern "C" void kernel_launch(void* const* d_in, const int* in_sizes, int n_in,
                              void* d_out, int out_size, void* d_ws, size_t ws_size,
                              hipStream_t stream) {
  (void)in_sizes; (void)n_in; (void)out_size; (void)ws_size;
  const float* x       = (const float*)d_in[0];
  const float* mask    = (const float*)d_in[1];
  const float* cw1     = (const float*)d_in[2];
  const float* cb1     = (const float*)d_in[3];
  const float* dal_q   = (const float*)d_in[4];
  const float* dal_k   = (const float*)d_in[5];
  const float* dal_v   = (const float*)d_in[6];
  const float* rel_t   = (const float*)d_in[7];
  const float* blk_w   = (const float*)d_in[8];
  const float* blk_b   = (const float*)d_in[9];
  const float* cow     = (const float*)d_in[10];
  const float* cob     = (const float*)d_in[11];
  const float* tokens  = (const float*)d_in[12];
  const float* sm_in_w = (const float*)d_in[13];
  const float* sm_in_b = (const float*)d_in[14];
  const float* sm_out_w= (const float*)d_in[15];
  const float* sm_out_b= (const float*)d_in[16];
  const float* ca_in_w = (const float*)d_in[17];
  const float* ca_out_w= (const float*)d_in[18];

  char* ws = (char*)d_ws;
  size_t off = 0;
  auto alloc = [&](size_t bytes) -> void* {
    void* p = ws + off;
    off += (bytes + 255) & ~(size_t)255;
    return p;
  };

  // bf16 weight copies
  __bf16* cw1b  = (__bf16*)alloc((size_t)FM_ * DIM_ * 2);
  __bf16* dalqb = (__bf16*)alloc((size_t)L_ * FM_ * FM_ * 2);
  __bf16* dalkb = (__bf16*)alloc((size_t)L_ * FM_ * FM_ * 2);
  __bf16* dalvb = (__bf16*)alloc((size_t)L_ * FM_ * FM_ * 2);
  __bf16* blkb  = (__bf16*)alloc((size_t)L_ * FM_ * FM_ * 2);
  __bf16* smkvb = (__bf16*)alloc((size_t)2 * FM_ * FM_ * 2);   // sm_in_w rows 512..1535
  __bf16* caqb  = (__bf16*)alloc((size_t)FM_ * FM_ * 2);       // ca_in_w rows 0..511
  __bf16* caob  = (__bf16*)alloc((size_t)FM_ * FM_ * 2);
  __bf16* cowb  = (__bf16*)alloc((size_t)NC_ * FM_ * 2);
  // activations
  __bf16* xb    = (__bf16*)alloc((size_t)B_ * DIM_ * T_ * 2);
  float*  outf  = (float*) alloc((size_t)B_ * FM_ * T_ * 4);
  __bf16* outb  = (__bf16*)alloc((size_t)B_ * FM_ * T_ * 2);
  __bf16* kvseq = (__bf16*)alloc((size_t)B_ * 2 * FM_ * T_ * 2);   // later reused as qca
  __bf16* qkv   = (__bf16*)alloc((size_t)B_ * 3 * FM_ * T_ * 2);   // later reused as oca
  __bf16* hb    = (__bf16*)alloc((size_t)B_ * FM_ * T_ * 2);
  float*  qsm   = (float*) alloc((size_t)NT_ * FM_ * 4);
  float*  osm   = (float*) alloc((size_t)B_ * NT_ * FM_ * 4);
  float*  summary = (float*)alloc((size_t)B_ * NT_ * FM_ * 4);
  float*  kvca  = (float*) alloc((size_t)B_ * NT_ * 2 * FM_ * 4);

  auto conv = [&](const float* s, __bf16* d, size_t n) {
    f32_to_bf16_k<<<dim3((unsigned)((n + 255) / 256)), dim3(256), 0, stream>>>(s, d, n);
  };
  conv(cw1,   cw1b,  (size_t)FM_ * DIM_);
  conv(dal_q, dalqb, (size_t)L_ * FM_ * FM_);
  conv(dal_k, dalkb, (size_t)L_ * FM_ * FM_);
  conv(dal_v, dalvb, (size_t)L_ * FM_ * FM_);
  conv(blk_w, blkb,  (size_t)L_ * FM_ * FM_);
  conv(sm_in_w + (size_t)FM_ * FM_, smkvb, (size_t)2 * FM_ * FM_);
  conv(ca_in_w, caqb, (size_t)FM_ * FM_);
  conv(ca_out_w, caob, (size_t)FM_ * FM_);
  conv(cow, cowb, (size_t)NC_ * FM_);
  conv(x, xb, (size_t)B_ * DIM_ * T_);

  // q projection of summary tokens (batch-independent, layer-independent)
  qsm_proj_k<<<dim3(NT_), dim3(256), 0, stream>>>(tokens, sm_in_w, sm_in_b, qsm);

  const dim3 blk256(256);
  const dim3 gFM (T_ / BN, FM_ / BM, B_);        // M=512
  const dim3 g2FM(T_ / BN, (2 * FM_) / BM, B_);  // M=1024
  const dim3 gNC (T_ / BN, (NC_ + BM - 1) / BM, B_);

  // out = cw1 @ x + cb1
  wmma_gemm<true,false,false,true,true><<<gFM, blk256, 0, stream>>>(
      cw1b, xb, cb1, nullptr, nullptr, outf, outb, FM_, DIM_, T_, FM_, 0);

  for (int i = 0; i < L_; ++i) {
    // --- summary branch (uses out at loop entry) ---
    wmma_gemm<true,false,false,false,true><<<g2FM, blk256, 0, stream>>>(
        smkvb, outb, sm_in_b + FM_, nullptr, nullptr, nullptr, kvseq,
        2 * FM_, FM_, T_, 2 * FM_, 0);
    sm_attention_k<<<dim3(NT_, NH_, B_), blk256, 0, stream>>>(qsm, kvseq, osm);
    smout_acc_k<<<dim3(B_ * NT_), blk256, 0, stream>>>(osm, sm_out_w, sm_out_b, summary, i == 0);

    // --- DAL q/k/v projections into one (B,1536,T) buffer ---
    wmma_gemm<false,false,false,false,true><<<gFM, blk256, 0, stream>>>(
        dalqb + (size_t)i * FM_ * FM_, outb, nullptr, nullptr, nullptr, nullptr, qkv,
        FM_, FM_, T_, 3 * FM_, 0);
    wmma_gemm<false,false,false,false,true><<<gFM, blk256, 0, stream>>>(
        dalkb + (size_t)i * FM_ * FM_, outb, nullptr, nullptr, nullptr, nullptr, qkv,
        FM_, FM_, T_, 3 * FM_, FM_);
    wmma_gemm<false,false,false,false,true><<<gFM, blk256, 0, stream>>>(
        dalvb + (size_t)i * FM_ * FM_, outb, nullptr, nullptr, nullptr, nullptr, qkv,
        FM_, FM_, T_, 3 * FM_, 2 * FM_);
    dal_elem_k<<<dim3((unsigned)(((size_t)B_ * FM_ * T_) / 256)), blk256, 0, stream>>>(
        qkv, rel_t + (size_t)i * FM_ * 3, hb, 1 << i);

    // --- out = (out + blk_w @ h + blk_b) * mask ---
    wmma_gemm<true,true,true,true,true><<<gFM, blk256, 0, stream>>>(
        blkb + (size_t)i * FM_ * FM_, hb, blk_b + (size_t)i * FM_, outf, mask,
        outf, outb, FM_, FM_, T_, FM_, 0);
  }

  // --- cross attention: q from out, k/v from summary/5 ---
  __bf16* qca = kvseq;  // reuse
  __bf16* oca = qkv;    // reuse
  wmma_gemm<false,false,false,false,true><<<gFM, blk256, 0, stream>>>(
      caqb, outb, nullptr, nullptr, nullptr, nullptr, qca, FM_, FM_, T_, FM_, 0);
  ca_kv_k<<<dim3(B_ * NT_), blk256, 0, stream>>>(summary, ca_in_w + (size_t)FM_ * FM_, kvca);
  ca_att_k<<<dim3(T_ / 64, B_), blk256, 0, stream>>>(qca, kvca, oca);
  // out = ca_out_w @ oca + out
  wmma_gemm<false,true,false,true,true><<<gFM, blk256, 0, stream>>>(
      caob, oca, nullptr, outf, nullptr, outf, outb, FM_, FM_, T_, FM_, 0);

  // --- final classifier: d_out = (cow @ out + cob) * mask ---
  wmma_gemm<true,false,true,true,false><<<gNC, blk256, 0, stream>>>(
      cowb, outb, cob, nullptr, mask, (float*)d_out, nullptr, NC_, FM_, T_, NC_, 0);
}
